// FastAttention_42511586296582
// MI455X (gfx1250) — compile-verified, matrix operand order checked by reading, not run on hardware
//
#include <hip/hip_runtime.h>
#include <hip/hip_bf16.h>
#include <math.h>

// ---------------------------------------------------------------------------
// FastAttention on MI455X (gfx1250, wave32, WMMA).
//
// Structure (traced from the reference's padding/index arithmetic):
//   1) Q/K/V = X @ W^T + b : 4096x512 @ 512x512 GEMMs -> v_wmma_f32_16x16x32_f16
//      (Q additionally scaled by 1/(8*exp(beta[head])) per column).
//   2) Dilation 0 collapses to: sig = sigmoid(q_s . k_s);  k2,v2 = sum_s sig*(k,v)
//   3) Dilation 1 collapses to: l_b = q[64b+63] . k2 ; p = softmax([l_0..l_31, 0])
//      attn_out nonzero only at last 32 tokens: coef_b * v2 (coef_31 += 1).
//   4) out = attn_out @ Wo^T + bo : rows < 2016 are exactly bo (broadcast fill);
//      tail is a 64x512 @ 512x512 WMMA GEMM.
//
// GEMM tiling: each wave computes a 16x64 strip (4 accumulators); A fragment
// reused across 4 WMMAs; the 8 waves of a block cover the 8 column-quads of
// one 16-row strip (identical A addresses -> WGP$ hits).  Fragments are
// explicitly double-buffered so next-step loads issue before current-step
// WMMAs (restores nonzero s_wait_loadcnt overlap inside a wave).
// ---------------------------------------------------------------------------

typedef __attribute__((ext_vector_type(16))) _Float16 v16h;
typedef __attribute__((ext_vector_type(8)))  _Float16 v8h;
typedef __attribute__((ext_vector_type(8)))  float    v8f;

static constexpr int SRC_LEN = 2048;
static constexpr int BSZ     = 2;
static constexpr int DM      = 512;
static constexpr int NTOK    = SRC_LEN * BSZ;   // 4096 rows
static constexpr int DH      = 64;

__device__ __forceinline__ v16h cat8(v8h lo, v8h hi) {
  v16h r;
#pragma unroll
  for (int i = 0; i < 8; ++i) { r[i] = lo[i]; r[i + 8] = hi[i]; }
  return r;
}

__device__ __forceinline__ v16h load_frag(const _Float16* p) {
  return cat8(*(const v8h*)(p), *(const v8h*)(p + 16));
}

// One wave: 16x64 f32 strip of A(MxK) @ W^T (W row-major; rows = out columns).
// K must be a multiple of 32 (compile-time constant at call sites -> unrolled).
// Double-buffered: step k+1's 1 A + 4 B fragment loads are issued before the
// 4 WMMAs of step k.
__device__ __forceinline__ void wmma_gemm_strip4(const _Float16* __restrict__ A,
                                                 const _Float16* __restrict__ W,
                                                 int m0, int n0, int K, int lda,
                                                 int ldw, v8f c[4]) {
  const int lane = threadIdx.x & 31;
  const int half = lane >> 4;                       // 0 or 1
  const _Float16* pa  = A + (size_t)(m0 + (lane & 15)) * lda + half * 8;
  const _Float16* pw0 = W + (size_t)(n0 + (lane & 15)) * ldw + half * 8;

  v16h a = load_frag(pa);
  v16h b[4];
#pragma unroll
  for (int t = 0; t < 4; ++t) b[t] = load_frag(pw0 + (size_t)(t * 16) * ldw);

#pragma unroll
  for (int k0 = 0; k0 < 512 - 32; k0 += 32) {       // K == 512 at all call sites
    const int k1 = k0 + 32;
    v16h a_n = load_frag(pa + k1);
    v16h b_n[4];
#pragma unroll
    for (int t = 0; t < 4; ++t)
      b_n[t] = load_frag(pw0 + (size_t)(t * 16) * ldw + k1);
#pragma unroll
    for (int t = 0; t < 4; ++t)
      c[t] = __builtin_amdgcn_wmma_f32_16x16x32_f16(false, a, false, b[t],
                                                    (short)0, c[t], false, false);
    a = a_n;
#pragma unroll
    for (int t = 0; t < 4; ++t) b[t] = b_n[t];
  }
#pragma unroll
  for (int t = 0; t < 4; ++t)
    c[t] = __builtin_amdgcn_wmma_f32_16x16x32_f16(false, a, false, b[t],
                                                  (short)0, c[t], false, false);
}

// ---------------------------------------------------------------------------
// Kernel 1: f32 -> f16 conversion of X and the four weight matrices.
__global__ void __launch_bounds__(256)
prep_kernel(const float* __restrict__ x, const float* __restrict__ Wq,
            const float* __restrict__ Wk, const float* __restrict__ Wv,
            const float* __restrict__ Wo, _Float16* __restrict__ Xh,
            _Float16* __restrict__ Wqh, _Float16* __restrict__ Wkh,
            _Float16* __restrict__ Wvh, _Float16* __restrict__ Woh) {
  const int NX = NTOK * DM;          // 2,097,152
  const int NW = DM * DM;            // 262,144
  const int total = NX + 4 * NW;
  for (int i = blockIdx.x * blockDim.x + threadIdx.x; i < total;
       i += gridDim.x * blockDim.x) {
    if (i < NX) {
      Xh[i] = (_Float16)x[i];
    } else {
      int j = i - NX;
      int wsel = j / NW, r = j - wsel * NW;
      const float* src = (wsel == 0) ? Wq : (wsel == 1) ? Wk : (wsel == 2) ? Wv : Wo;
      _Float16* dst = (wsel == 0) ? Wqh : (wsel == 1) ? Wkh : (wsel == 2) ? Wvh : Woh;
      dst[r] = (_Float16)src[r];
    }
  }
}

// ---------------------------------------------------------------------------
// Kernel 2: broadcast bo into all output rows for tokens t < 2016.
__global__ void __launch_bounds__(256)
fill_bo_kernel(const float* __restrict__ bo, float* __restrict__ out) {
  const int N = (SRC_LEN - 32) * BSZ * DM;   // 4032 rows * 512
  for (int i = blockIdx.x * blockDim.x + threadIdx.x; i < N;
       i += gridDim.x * blockDim.x)
    out[i] = bo[i & (DM - 1)];
}

// ---------------------------------------------------------------------------
// Kernel 3: Q/K/V projections.  Block = (matrix, 16-row strip); the 8 waves
// cover the 8 column-quads (n0 = wave*64), sharing identical A-fragment loads.
__global__ void __launch_bounds__(256)
gemm_qkv_kernel(const _Float16* __restrict__ Xh, const _Float16* __restrict__ Wqh,
                const _Float16* __restrict__ Wkh, const _Float16* __restrict__ Wvh,
                const float* __restrict__ bq, const float* __restrict__ bk,
                const float* __restrict__ bv, const float* __restrict__ beta,
                float* __restrict__ Q, float* __restrict__ K,
                float* __restrict__ V) {
  const int mat = blockIdx.x >> 8;                 // 0=Q 1=K 2=V
  const int m0  = (blockIdx.x & 255) << 4;         // 256 row strips
  const int n0  = (threadIdx.x >> 5) << 6;         // wave -> 64-col quad
  const _Float16* W = (mat == 0) ? Wqh : (mat == 1) ? Wkh : Wvh;
  const float* bias = (mat == 0) ? bq : (mat == 1) ? bk : bv;
  float* Out        = (mat == 0) ? Q : (mat == 1) ? K : V;

  v8f c[4] = {{}, {}, {}, {}};
  wmma_gemm_strip4(Xh, W, m0, n0, DM, DM, DM, c);

  const int lane  = threadIdx.x & 31;
  const int rbase = m0 + ((lane >> 4) << 3);
#pragma unroll
  for (int t = 0; t < 4; ++t) {
    const int col  = n0 + t * 16 + (lane & 15);
    const float bi = bias[col];
    float scale = 1.0f;
    if (mat == 0) scale = 0.125f * expf(-beta[col >> 6]);  // 1/(sqrt(64)*e^beta)
#pragma unroll
    for (int r = 0; r < 8; ++r)
      Out[(size_t)(rbase + r) * DM + col] = (c[t][r] + bi) * scale;
  }
}

// ---------------------------------------------------------------------------
// Kernel 4: dilation-0 collapse. One block per z; per-position sigmoid(q.k)
// weighted sum of k and v over all 2048 positions -> k2[z], v2[z].
__global__ void __launch_bounds__(256)
attn0_kernel(const float* __restrict__ Q, const float* __restrict__ K,
             const float* __restrict__ V, float* __restrict__ k2,
             float* __restrict__ v2) {
  const int z = blockIdx.x;                  // z = batch*8 + head
  const int batch = z >> 3, head = z & 7;
  const int lane = threadIdx.x & 31;
  const int wv = threadIdx.x >> 5;           // 8 waves
  __shared__ float sK[8][DH];
  __shared__ float sV[8][DH];

  float ak0 = 0.f, ak1 = 0.f, av0 = 0.f, av1 = 0.f;
  const int colbase = head * DH + lane * 2;  // 2 dims per lane
  for (int s = wv; s < SRC_LEN; s += 8) {
    const size_t row = (size_t)(s * BSZ + batch) * DM + colbase;
    const float2 qv = *(const float2*)(Q + row);
    const float2 kv = *(const float2*)(K + row);
    const float2 vv = *(const float2*)(V + row);
    float p = qv.x * kv.x + qv.y * kv.y;
#pragma unroll
    for (int off = 16; off > 0; off >>= 1) p += __shfl_xor(p, off, 32);
    const float sg = 1.0f / (1.0f + expf(-p));   // softmax([w,0]) == sigmoid
    ak0 += sg * kv.x; ak1 += sg * kv.y;
    av0 += sg * vv.x; av1 += sg * vv.y;
  }
  sK[wv][lane * 2] = ak0; sK[wv][lane * 2 + 1] = ak1;
  sV[wv][lane * 2] = av0; sV[wv][lane * 2 + 1] = av1;
  __syncthreads();
  if (threadIdx.x < DH) {
    float s = 0.f;
#pragma unroll
    for (int i = 0; i < 8; ++i) s += sK[i][threadIdx.x];
    k2[z * DH + threadIdx.x] = s;
  } else if (threadIdx.x < 2 * DH) {
    const int j = threadIdx.x - DH;
    float s = 0.f;
#pragma unroll
    for (int i = 0; i < 8; ++i) s += sV[i][j];
    v2[z * DH + j] = s;
  }
}

// ---------------------------------------------------------------------------
// Kernel 5: dilation-1 collapse + build f16 tail activations.
// lane b: l_b = q[z,64b+63].k2[z]; softmax over 32 lanes + sink(0);
// AOh[row=b*2+batch, head*64..] = coef_b * v2[z]  (coef_31 += 1 from dil 0).
__global__ void __launch_bounds__(32)
attn1_kernel(const float* __restrict__ Q, const float* __restrict__ k2,
             const float* __restrict__ v2, _Float16* __restrict__ AOh) {
  const int z = blockIdx.x;
  const int batch = z >> 3, head = z & 7;
  const int b = threadIdx.x;                         // 0..31
  const float* qp  = Q + (size_t)((64 * b + 63) * BSZ + batch) * DM + head * DH;
  const float* k2p = k2 + z * DH;
  float l = 0.f;
#pragma unroll
  for (int j = 0; j < DH; ++j) l += qp[j] * k2p[j];
  float m = l;
#pragma unroll
  for (int off = 16; off > 0; off >>= 1) m = fmaxf(m, __shfl_xor(m, off, 32));
  m = fmaxf(m, 0.0f);                                // attention sink logit 0
  const float e = expf(l - m);
  float sum = e;
#pragma unroll
  for (int off = 16; off > 0; off >>= 1) sum += __shfl_xor(sum, off, 32);
  sum += expf(-m);                                   // sink term
  const float coef = e / sum + ((b == 31) ? 1.0f : 0.0f);
  const float* v2p = v2 + z * DH;
  _Float16* dst = AOh + (size_t)(b * BSZ + batch) * DM + head * DH;
#pragma unroll
  for (int j = 0; j < DH; ++j) dst[j] = (_Float16)(coef * v2p[j]);
}

// ---------------------------------------------------------------------------
// Kernel 6: tail output GEMM (64x512 @ 512x512) + bo, into last 32 tokens.
__global__ void __launch_bounds__(256)
gemm_out_kernel(const _Float16* __restrict__ AOh, const _Float16* __restrict__ Woh,
                const float* __restrict__ bo, float* __restrict__ out) {
  const int m0 = blockIdx.x << 4;                  // 4 row strips (64 rows)
  const int n0 = (threadIdx.x >> 5) << 6;          // wave -> 64-col quad
  v8f c[4] = {{}, {}, {}, {}};
  wmma_gemm_strip4(AOh, Woh, m0, n0, DM, DM, DM, c);
  const int lane  = threadIdx.x & 31;
  const int rbase = m0 + ((lane >> 4) << 3);
  const int rowoff = (SRC_LEN - 32) * BSZ;         // 4032
#pragma unroll
  for (int t = 0; t < 4; ++t) {
    const int col  = n0 + t * 16 + (lane & 15);
    const float bi = bo[col];
#pragma unroll
    for (int r = 0; r < 8; ++r)
      out[(size_t)(rowoff + rbase + r) * DM + col] = c[t][r] + bi;
  }
}

// ---------------------------------------------------------------------------
extern "C" void kernel_launch(void* const* d_in, const int* in_sizes, int n_in,
                              void* d_out, int out_size, void* d_ws, size_t ws_size,
                              hipStream_t stream) {
  const float* x    = (const float*)d_in[0];
  const float* Wq   = (const float*)d_in[1];
  const float* bq   = (const float*)d_in[2];
  const float* Wk   = (const float*)d_in[3];
  const float* bk   = (const float*)d_in[4];
  const float* Wv   = (const float*)d_in[5];
  const float* bv   = (const float*)d_in[6];
  const float* Wo   = (const float*)d_in[7];
  const float* bo   = (const float*)d_in[8];
  const float* beta = (const float*)d_in[9];
  float* out = (float*)d_out;
  char*  ws  = (char*)d_ws;

  // Workspace layout (~31.5 MB)
  const size_t szXh = (size_t)NTOK * DM * sizeof(_Float16);   // 4 MB
  const size_t szWh = (size_t)DM * DM * sizeof(_Float16);     // 512 KB
  const size_t szQ  = (size_t)NTOK * DM * sizeof(float);      // 8 MB
  _Float16* Xh  = (_Float16*)(ws);
  _Float16* Wqh = (_Float16*)(ws + szXh);
  _Float16* Wkh = (_Float16*)(ws + szXh + szWh);
  _Float16* Wvh = (_Float16*)(ws + szXh + 2 * szWh);
  _Float16* Woh = (_Float16*)(ws + szXh + 3 * szWh);
  float* Q  = (float*)(ws + szXh + 4 * szWh);
  float* K  = (float*)(ws + szXh + 4 * szWh + szQ);
  float* V  = (float*)(ws + szXh + 4 * szWh + 2 * szQ);
  float* k2 = (float*)(ws + szXh + 4 * szWh + 3 * szQ);
  float* v2 = k2 + 16 * DH;
  _Float16* AOh = (_Float16*)(v2 + 16 * DH);                  // 64x512 f16

  // 1) f32->f16 conversions (independent of 2)
  prep_kernel<<<2048, 256, 0, stream>>>(x, Wq, Wk, Wv, Wo, Xh, Wqh, Wkh, Wvh, Woh);
  // 2) bo broadcast for tokens < 2016
  fill_bo_kernel<<<1024, 256, 0, stream>>>(bo, out);
  // 3) QKV projection GEMMs: 3 mats x 256 row strips, 8 column-quad waves
  gemm_qkv_kernel<<<768, 256, 0, stream>>>(Xh, Wqh, Wkh, Wvh, bq, bk, bv, beta,
                                           Q, K, V);
  // 4) dilation-0 sigmoid-weighted reduction -> k2, v2
  attn0_kernel<<<16, 256, 0, stream>>>(Q, K, V, k2, v2);
  // 5) dilation-1 softmax(+sink) -> f16 tail activations
  attn1_kernel<<<16, 32, 0, stream>>>(Q, k2, v2, AOh);
  // 6) tail output GEMM + bo
  gemm_out_kernel<<<4, 256, 0, stream>>>(AOh, Woh, bo, out);
}
